// PilotNet_38560216383513
// MI455X (gfx1250) — 8-bit matrix operand order repaired and checked by reading, compiled, not run on hardware
//
#include <hip/hip_runtime.h>


typedef __attribute__((ext_vector_type(8))) int v8i;

__device__ __forceinline__ float clampf(float v, float lo, float hi) {
  return fminf(fmaxf(v, lo), hi);
}


__global__ void k_init_slots(float* slots) {
  if (threadIdx.x < 64) slots[threadIdx.x] = 0.0f;
}

__global__ void k_absmax(const float* __restrict__ x, int n,
                         unsigned* __restrict__ slot, int use_abs) {
  __shared__ float red[256];
  int tid = threadIdx.x;
  float m = 0.0f;
  for (int i = blockIdx.x * 256 + tid; i < n; i += gridDim.x * 256) {
    float v = x[i];
    if (use_abs) v = fabsf(v);
    m = fmaxf(m, v);
  }
  red[tid] = m;
  __syncthreads();
  for (int s = 128; s > 0; s >>= 1) {
    if (tid < s) red[tid] = fmaxf(red[tid], red[tid + s]);
    __syncthreads();
  }
  if (tid == 0) atomicMax(slot, __float_as_uint(red[0]));
}

__global__ void k_quant_w(const float* __restrict__ w,
                          signed char* __restrict__ codes,
                          const float* __restrict__ maxabs_slot,
                          int O, int K, int K_pad, int O_pad) {
  int idx = blockIdx.x * 256 + threadIdx.x;
  int total = O_pad * K_pad;
  if (idx >= total) return;
  int o = idx / K_pad, k = idx - o * K_pad;
  float s = fmaxf(maxabs_slot[0], 1e-8f) / 7.0f;
  signed char c = 0;
  if (o < O && k < K)
    c = (signed char)(int)rintf(clampf(w[(long long)o * K + k] / s, -7.0f, 7.0f));
  codes[idx] = c;
}

__global__ void k_dequant_b(const float* __restrict__ b, float* __restrict__ bdq,
                            const float* __restrict__ maxabs_slot, int n) {
  int i = blockIdx.x * 256 + threadIdx.x;
  if (i >= n) return;
  float s = fmaxf(maxabs_slot[0], 1e-8f) / 127.0f;
  bdq[i] = rintf(clampf(b[i] / s, -127.0f, 127.0f)) * s;
}

__global__ void k_quant_x(const float* __restrict__ x,
                          unsigned char* __restrict__ codes,
                          const float* __restrict__ max_slot, int n) {
  int i = blockIdx.x * 256 + threadIdx.x;
  if (i >= n) return;
  float s = fmaxf(max_slot[0] / 255.0f, 1e-8f) / 255.0f;
  codes[i] = (unsigned char)(int)rintf(clampf((x[i] / 255.0f) / s, 0.0f, 255.0f));
}

__global__ void k_quant_act(const float* __restrict__ raw,
                            unsigned char* __restrict__ codes,
                            const float* __restrict__ max_slot, int n) {
  int i = blockIdx.x * 256 + threadIdx.x;
  if (i >= n) return;
  float s = fmaxf(max_slot[0], 1e-8f) / 15.0f;
  codes[i] = (unsigned char)(int)rintf(clampf(raw[i] / s, 0.0f, 15.0f));
}

__global__ void k_quant_act_pad(const float* __restrict__ raw,
                                unsigned char* __restrict__ codes,
                                const float* __restrict__ max_slot,
                                int rows, int O, int K_pad) {
  int idx = blockIdx.x * 256 + threadIdx.x;
  if (idx >= rows * K_pad) return;
  int r = idx / K_pad, c = idx - r * K_pad;
  float s = fmaxf(max_slot[0], 1e-8f) / 15.0f;
  unsigned char v = 0;
  if (c < O) v = (unsigned char)(int)rintf(clampf(raw[r * O + c] / s, 0.0f, 15.0f));
  codes[idx] = v;
}

template <int KH, int KW, int STRIDE>
__global__ __launch_bounds__(256) void k_conv_wmma(
    const unsigned char* __restrict__ in_codes,
    const signed char* __restrict__ w_codes,
    const float* __restrict__ bias_dq,
    const float* __restrict__ bn_g, const float* __restrict__ bn_b,
    const float* __restrict__ bn_rm, const float* __restrict__ bn_rv,
    float* __restrict__ raw, unsigned* __restrict__ act_max,
    const float* __restrict__ sa_slot, float pre_div, float qn,
    const float* __restrict__ wmax_slot,
    int C, int H, int W, int O, int Ho, int Wo,
    int K, int K_pad, int P) {
  constexpr int KHW = KH * KW;
  __shared__ __align__(16) unsigned char atile[2][1024];
  __shared__ float red[256];
  const int tid = threadIdx.x, wave = tid >> 5, lane = tid & 31;
  const int mt = wave >> 2;
  const int nt = wave & 3;
  const int img = blockIdx.z;
  const int block_pix = blockIdx.x * 32;
  const int pixel_base = block_pix + mt * 16;
  const int oc = blockIdx.y * 64 + nt * 16 + (lane & 15);
  const int hb8 = (lane >> 4) << 3;
  const int hb16 = (lane >> 4) << 4;
  const int HW = H * W;
  const long long in_base = (long long)img * C * HW;

  const int tile_i = tid >> 7;
  const int m_l = (tid & 127) >> 3;
  const int kk = (tid & 7) * 8;
  const int p_fill = block_pix + tile_i * 16 + m_l;
  const bool p_ok = p_fill < P;
  int oy = p_fill / Wo;
  int ox = p_fill - oy * Wo;
  const long long rowoff = in_base + (long long)(oy * STRIDE) * W + ox * STRIDE;
  uint2* const fill_dst = (uint2*)(&atile[0][0]) + tid;

  v8i acc = {};

  for (int k0 = 0; k0 < K_pad; k0 += 64) {
    int kb = k0 + kk;
    int c = kb / KHW;
    int r = kb - c * KHW;
    int kh = r / KW;
    int kw = r - kh * KW;
    long long off = rowoff + (long long)c * HW + (long long)kh * W;
    unsigned lo = 0, hi = 0;
    if (p_ok) {
      #pragma unroll
      for (int b = 0; b < 8; b++) {
        unsigned v = 0;
        if (kb + b < K) v = in_codes[off + kw];
        if (b < 4) lo |= v << (8 * b); else hi |= v << (8 * (b - 4));
        kw++;
        if (kw == KW) {
          kw = 0; off += W; kh++;
          if (kh == KH) { kh = 0; off += HW - KH * W; }
        }
      }
    }
    uint2 w2; w2.x = lo; w2.y = hi;
    *fill_dst = w2;
    __syncthreads();

    const unsigned char* ab = &atile[mt][(lane & 15) * 64 + hb16];
    uint2 a0 = *(const uint2*)(ab + 0);
    uint2 a1 = *(const uint2*)(ab + 8);
    uint2 a2 = *(const uint2*)(ab + 32);
    uint2 a3 = *(const uint2*)(ab + 40);
    const signed char* wb = w_codes + (long long)oc * K_pad + k0 + hb16;
    uint4 b0 = *(const uint4*)(wb);
    uint4 b1 = *(const uint4*)(wb + 32);
    __syncthreads();

    v8i A, Bm;
    A[0] = (int)a0.x; A[1] = (int)a0.y; A[2] = (int)a1.x; A[3] = (int)a1.y;
    A[4] = (int)a2.x; A[5] = (int)a2.y; A[6] = (int)a3.x; A[7] = (int)a3.y;
    Bm[0] = (int)b0.x; Bm[1] = (int)b0.y; Bm[2] = (int)b0.z; Bm[3] = (int)b0.w;
    Bm[4] = (int)b1.x; Bm[5] = (int)b1.y; Bm[6] = (int)b1.z; Bm[7] = (int)b1.w;
    acc = __builtin_amdgcn_wmma_i32_16x16x64_iu8(false, A, true, Bm, acc,
                                                 false, false);
  }

  float sa = fmaxf(sa_slot[0] / pre_div, 1e-8f) / qn;
  float sw = fmaxf(wmax_slot[0], 1e-8f) / 7.0f;
  float sc = sa * sw;
  float lmax = 0.0f;
  if (oc < O) {
    float bns = bn_g[oc] * rsqrtf(bn_rv[oc] + 1e-5f);
    float bdq = bias_dq[oc];
    float mu = bn_rm[oc], bet = bn_b[oc];
    long long obase = ((long long)img * O + oc) * P;
    #pragma unroll
    for (int r8 = 0; r8 < 8; r8++) {
      int p = pixel_base + r8 + hb8;
      if (p < P) {
        float v = (float)acc[r8] * sc + bdq;
        v = (v - mu) * bns + bet;
        v = fmaxf(v, 0.0f);
        raw[obase + p] = v;
        lmax = fmaxf(lmax, v);
      }
    }
  }
  red[tid] = lmax;
  __syncthreads();
  for (int s = 128; s > 0; s >>= 1) {
    if (tid < s) red[tid] = fmaxf(red[tid], red[tid + s]);
    __syncthreads();
  }
  if (tid == 0) atomicMax(act_max, __float_as_uint(red[0]));
}

__global__ __launch_bounds__(256) void k_fc_wmma(
    const unsigned char* __restrict__ a_codes,
    const signed char* __restrict__ w_codes,
    const float* __restrict__ bias_dq,
    float* __restrict__ raw, unsigned* __restrict__ act_max,
    const float* __restrict__ sa_slot,
    const float* __restrict__ wmax_slot,
    int O, int K_pad) {
  __shared__ float red[256];
  const int tid = threadIdx.x, wave = tid >> 5, lane = tid & 31;
  const int row_base = (blockIdx.x * 8 + wave) * 16;
  const int oc = blockIdx.y * 16 + (lane & 15);
  const int hb8 = (lane >> 4) << 3;
  const int hb16 = (lane >> 4) << 4;
  const int arow = row_base + (lane & 15);

  v8i acc = {};
  for (int k0 = 0; k0 < K_pad; k0 += 64) {
    const unsigned char* ab = a_codes + (long long)arow * K_pad + k0 + hb16;
    uint2 a0 = *(const uint2*)(ab + 0);
    uint2 a1 = *(const uint2*)(ab + 8);
    uint2 a2 = *(const uint2*)(ab + 32);
    uint2 a3 = *(const uint2*)(ab + 40);
    const signed char* wb = w_codes + (long long)oc * K_pad + k0 + hb16;
    uint4 b0 = *(const uint4*)(wb);
    uint4 b1 = *(const uint4*)(wb + 32);
    v8i A, Bm;
    A[0] = (int)a0.x; A[1] = (int)a0.y; A[2] = (int)a1.x; A[3] = (int)a1.y;
    A[4] = (int)a2.x; A[5] = (int)a2.y; A[6] = (int)a3.x; A[7] = (int)a3.y;
    Bm[0] = (int)b0.x; Bm[1] = (int)b0.y; Bm[2] = (int)b0.z; Bm[3] = (int)b0.w;
    Bm[4] = (int)b1.x; Bm[5] = (int)b1.y; Bm[6] = (int)b1.z; Bm[7] = (int)b1.w;
    acc = __builtin_amdgcn_wmma_i32_16x16x64_iu8(false, A, true, Bm, acc,
                                                 false, false);
  }

  float sa = fmaxf(sa_slot[0], 1e-8f) / 15.0f;
  float sw = fmaxf(wmax_slot[0], 1e-8f) / 7.0f;
  float sc = sa * sw;
  float lmax = 0.0f;
  if (oc < O) {
    float bdq = bias_dq[oc];
    #pragma unroll
    for (int r = 0; r < 8; r++) {
      int rw = row_base + r + hb8;
      float v = fmaxf((float)acc[r] * sc + bdq, 0.0f);
      raw[(long long)rw * O + oc] = v;
      lmax = fmaxf(lmax, v);
    }
  }
  red[tid] = lmax;
  __syncthreads();
  for (int s = 128; s > 0; s >>= 1) {
    if (tid < s) red[tid] = fmaxf(red[tid], red[tid + s]);
    __syncthreads();
  }
  if (tid == 0) atomicMax(act_max, __float_as_uint(red[0]));
}

__global__ void k_final(const unsigned char* __restrict__ codes3,
                        const float* __restrict__ s3_slot,
                        const float* __restrict__ ow, const float* __restrict__ ob,
                        const float* __restrict__ owmax_slot,
                        const float* __restrict__ obmax_slot,
                        float* __restrict__ out) {
  int i = blockIdx.x * 256 + threadIdx.x;
  if (i >= 256) return;
  float s3 = fmaxf(s3_slot[0], 1e-8f) / 15.0f;
  float sow = fmaxf(owmax_slot[0], 1e-8f) / 7.0f;
  float sob = fmaxf(obmax_slot[0], 1e-8f) / 127.0f;
  float obq = rintf(clampf(ob[0] / sob, -127.0f, 127.0f)) * sob;
  float acc = 0.0f;
  #pragma unroll
  for (int j = 0; j < 10; j++) {
    float wq = rintf(clampf(ow[j] / sow, -7.0f, 7.0f));
    acc += (float)codes3[i * 16 + j] * wq;
  }
  float y = acc * s3 * sow + obq;
  y = clampf(y, -1.0f, 1.0f);
  out[i] = rintf(y * 16384.0f) / 16384.0f;
}


struct ConvDesc { int C, H, W, O, KH, KW, stride, Ho, Wo, O_pad, K, K_pad; };
struct FcDesc { int O, K, O_pad, K_pad; };

static inline int cdiv(long long a, long long b) { return (int)((a + b - 1) / b); }

extern "C" void kernel_launch(void* const* d_in, const int* in_sizes, int n_in,
                              void* d_out, int out_size, void* d_ws, size_t ws_size,
                              hipStream_t stream) {
  (void)in_sizes; (void)n_in; (void)out_size; (void)ws_size;
  const int B = 256;

  const float* x = (const float*)d_in[0];
  const float *cw[5], *cb[5], *gg[5], *bb[5], *rm[5], *rv[5];
  for (int i = 0; i < 5; i++) {
    int base = 1 + i * 6;
    cw[i] = (const float*)d_in[base + 0];
    cb[i] = (const float*)d_in[base + 1];
    gg[i] = (const float*)d_in[base + 2];
    bb[i] = (const float*)d_in[base + 3];
    rm[i] = (const float*)d_in[base + 4];
    rv[i] = (const float*)d_in[base + 5];
  }
  const float* fw[3] = {(const float*)d_in[31], (const float*)d_in[33], (const float*)d_in[35]};
  const float* fb[3] = {(const float*)d_in[32], (const float*)d_in[34], (const float*)d_in[36]};
  const float* ow = (const float*)d_in[37];
  const float* ob = (const float*)d_in[38];
  float* out = (float*)d_out;

  static const ConvDesc CD[5] = {
      {1, 240, 320, 24, 5, 5, 2, 118, 158, 64, 25, 64},
      {24, 118, 158, 36, 5, 5, 2, 57, 77, 64, 600, 640},
      {36, 57, 77, 48, 5, 5, 2, 27, 37, 64, 900, 960},
      {48, 27, 37, 64, 3, 3, 1, 25, 35, 64, 432, 448},
      {64, 25, 35, 64, 3, 3, 1, 23, 33, 64, 576, 576},
  };
  static const FcDesc FD[3] = {
      {100, 48576, 112, 48576}, {50, 100, 64, 128}, {10, 50, 16, 64}};
  static const int fc_out_kpad[3] = {128, 64, 16};
  static const int boff[5] = {0, 32, 80, 128, 192};
  static const int fboff[3] = {256, 360, 412};

  char* ws = (char*)d_ws;
  size_t cur = 0;
  auto alloc = [&](size_t bytes) -> char* {
    char* p = ws + cur;
    cur = (cur + bytes + 255) & ~(size_t)255;
    return p;
  };
  float* slots = (float*)alloc(256);
  unsigned* uslots = (unsigned*)slots;
  float* bias_dq = (float*)alloc(2048);
  signed char* wc[5];
  for (int i = 0; i < 5; i++) wc[i] = (signed char*)alloc((size_t)CD[i].O_pad * CD[i].K_pad);
  signed char* fwc[3];
  for (int i = 0; i < 3; i++) fwc[i] = (signed char*)alloc((size_t)FD[i].O_pad * FD[i].K_pad);
  unsigned char* codesA = (unsigned char*)alloc(40449024);
  unsigned char* codesB = (unsigned char*)alloc(114548736);
  float* raw = (float*)alloc((size_t)114548736 * 4);

  k_init_slots<<<1, 64, 0, stream>>>(slots);

  int nx = B * 1 * 240 * 320;
  k_absmax<<<2048, 256, 0, stream>>>(x, nx, uslots + 0, 0);
  k_quant_x<<<cdiv(nx, 256), 256, 0, stream>>>(x, codesA, slots + 0, nx);

  unsigned char* cur_in = codesA;
  unsigned char* nxt = codesB;

  for (int i = 0; i < 5; i++) {
    const ConvDesc& d = CD[i];
    int nw = d.O * d.K;
    k_absmax<<<cdiv(nw, 256) < 1024 ? cdiv(nw, 256) : 1024, 256, 0, stream>>>(
        cw[i], nw, uslots + 10 + i, 1);
    int wtot = d.O_pad * d.K_pad;
    k_quant_w<<<cdiv(wtot, 256), 256, 0, stream>>>(cw[i], wc[i], slots + 10 + i,
                                                   d.O, d.K, d.K_pad, d.O_pad);
    k_absmax<<<1, 256, 0, stream>>>(cb[i], d.O, uslots + 20 + i, 1);
    k_dequant_b<<<1, 256, 0, stream>>>(cb[i], bias_dq + boff[i], slots + 20 + i, d.O);

    int P = d.Ho * d.Wo;
    dim3 grid(cdiv(P, 32), d.O_pad / 64, B);
    float pre_div = (i == 0) ? 255.0f : 1.0f;
    float qn = (i == 0) ? 255.0f : 15.0f;
    const float* sa = slots + i;
    if (i < 3) {
      k_conv_wmma<5, 5, 2><<<grid, 256, 0, stream>>>(
          cur_in, wc[i], bias_dq + boff[i], gg[i], bb[i], rm[i], rv[i], raw,
          uslots + 1 + i, sa, pre_div, qn, slots + 10 + i, d.C, d.H, d.W, d.O,
          d.Ho, d.Wo, d.K, d.K_pad, P);
    } else {
      k_conv_wmma<3, 3, 1><<<grid, 256, 0, stream>>>(
          cur_in, wc[i], bias_dq + boff[i], gg[i], bb[i], rm[i], rv[i], raw,
          uslots + 1 + i, sa, pre_div, qn, slots + 10 + i, d.C, d.H, d.W, d.O,
          d.Ho, d.Wo, d.K, d.K_pad, P);
    }

    int na = B * d.O * P;
    k_quant_act<<<cdiv(na, 256), 256, 0, stream>>>(raw, nxt, slots + 1 + i, na);
    unsigned char* t = cur_in; cur_in = nxt; nxt = t;
  }

  for (int i = 0; i < 3; i++) {
    const FcDesc& d = FD[i];
    int nw = d.O * d.K;
    k_absmax<<<cdiv(nw, 256) < 1024 ? cdiv(nw, 256) : 1024, 256, 0, stream>>>(
        fw[i], nw, uslots + 15 + i, 1);
    int wtot = d.O_pad * d.K_pad;
    k_quant_w<<<cdiv(wtot, 256), 256, 0, stream>>>(fw[i], fwc[i], slots + 15 + i,
                                                   d.O, d.K, d.K_pad, d.O_pad);
    k_absmax<<<1, 256, 0, stream>>>(fb[i], d.O, uslots + 25 + i, 1);
    k_dequant_b<<<1, 256, 0, stream>>>(fb[i], bias_dq + fboff[i], slots + 25 + i, d.O);

    dim3 grid(2, d.O_pad / 16, 1);
    const float* sa = slots + 5 + i;
    k_fc_wmma<<<grid, 256, 0, stream>>>(cur_in, fwc[i], bias_dq + fboff[i], raw,
                                        uslots + 6 + i, sa, slots + 15 + i,
                                        d.O, d.K_pad);

    int np = B * fc_out_kpad[i];
    k_quant_act_pad<<<cdiv(np, 256), 256, 0, stream>>>(
        raw, nxt, slots + 6 + i, B, d.O, fc_out_kpad[i]);
    unsigned char* t = cur_in; cur_in = nxt; nxt = t;
  }

  k_absmax<<<1, 256, 0, stream>>>(ow, 10, uslots + 18, 1);
  k_absmax<<<1, 256, 0, stream>>>(ob, 1, uslots + 28, 1);
  k_final<<<1, 256, 0, stream>>>(cur_in, slots + 8, ow, ob, slots + 18,
                                 slots + 28, out);
}
